// Experts_84533546320328
// MI455X (gfx1250) — matrix operand order repaired and checked by reading, not compiled, not run on hardware
//
#include <hip/hip_runtime.h>

// Problem constants (match reference)
#define E_ 8
#define D_ 1024
#define F_ 4096
#define G_ 2048
#define BK 32

typedef __attribute__((ext_vector_type(16))) __bf16       v16bf;
typedef __attribute__((ext_vector_type(8)))  float        v8f;
typedef __attribute__((ext_vector_type(4)))  unsigned int v4u;

union FragU {
    struct { v4u lo, hi; } u;
    v16bf v;
};

static __device__ __forceinline__ unsigned short f2bf(float a) {
    union { float f; unsigned int u; } x{a};
    return (unsigned short)((x.u + 0x7fffu + ((x.u >> 16) & 1u)) >> 16);
}

static __device__ __forceinline__ unsigned int pack_bf16x2(float a, float b) {
    union { float f; unsigned int u; } x{a}, y{b};
    unsigned int ra = x.u + 0x7fffu + ((x.u >> 16) & 1u);
    unsigned int rb = y.u + 0x7fffu + ((y.u >> 16) & 1u);
    return (ra >> 16) | (rb & 0xffff0000u);
}

static __device__ __forceinline__ v8f wmma_bf16(v16bf a, v16bf b, v8f c) {
    // D = A(16x32 bf16) * B(32x16 bf16) + C(16x16 f32)
    return __builtin_amdgcn_wmma_f32_16x16x32_bf16(false, a, false, b, (short)0, c,
                                                   false, false);
}

// ---- CDNA5 async global->LDS copy (ASYNCcnt path) -----------------------
// Used only in the GEMM prologues: inside the K-loop the asm "memory"
// clobbers were observed to break accumulator register coalescing
// (v_mov_b64 storms around each WMMA), so the steady-state loop uses the
// plain global_load_b128 -> ds_store_b128 double-buffer pipeline instead.
static __device__ __forceinline__ unsigned int lds_off_of(const void* p) {
    // generic -> addrspace(3) cast yields the 32-bit LDS offset
    return (unsigned int)(unsigned long long)
        (__attribute__((address_space(3))) const void*)p;
}

static __device__ __forceinline__ void async_cp_b128(unsigned short* lds_dst,
                                                     const unsigned short* gsrc) {
    asm volatile("global_load_async_to_lds_b128 %0, %1, off"
                 :
                 : "v"(lds_off_of(lds_dst)), "v"(gsrc)
                 : "memory");
}

static __device__ __forceinline__ void wait_async0() {
    asm volatile("s_wait_asynccnt 0x0" ::: "memory");
}

// A fragment (16x32 bf16, M x K): lane&15 = M row, lane>>4 selects K-subgroup.
// VGPR halves 0-7 = K(kg*8 .. kg*8+7), halves 8-15 = K(16+kg*8 ..).
static __device__ __forceinline__ v16bf load_a_frag(const unsigned short* tile,
                                                    int ld, int lane) {
    const int r  = lane & 15;
    const int kg = lane >> 4;
    const unsigned short* p = tile + r * ld + kg * 8;
    FragU f;
    f.u.lo = *(const v4u*)(p);        // ds_load_b128
    f.u.hi = *(const v4u*)(p + 16);   // ds_load_b128
    return f.v;
}

// B fragment (32x16 bf16, K x N): lane&15 = N column, lane>>4 selects K-subgroup.
// halves 0-7 = K(kg*8 .. kg*8+7), halves 8-15 = K(16+kg*8 ..).
static __device__ __forceinline__ v16bf load_b_frag(const unsigned short* tile,
                                                    int ld, int lane) {
    const int n  = lane & 15;
    const int kg = lane >> 4;
    union { v16bf v; unsigned short s[16]; } t;
#pragma unroll
    for (int i = 0; i < 8; ++i) {
        t.s[i]     = tile[(kg * 8 + i) * ld + n];
        t.s[8 + i] = tile[(16 + kg * 8 + i) * ld + n];
    }
    return t.v;
}

// -------------------------------------------------------------------------
// fp32 -> bf16 streaming convert (8 elems / thread / iter)
// -------------------------------------------------------------------------
__global__ __launch_bounds__(256) void cvt_bf16(const float* __restrict__ in,
                                                unsigned short* __restrict__ out,
                                                long long n8) {
    long long i = (long long)blockIdx.x * blockDim.x + threadIdx.x;
    const long long stride = (long long)gridDim.x * blockDim.x;
    const float4* in4 = (const float4*)in;
    uint4* out4 = (uint4*)out;
    for (; i < n8; i += stride) {
        float4 a = in4[2 * i];
        float4 b = in4[2 * i + 1];
        uint4 o;
        o.x = pack_bf16x2(a.x, a.y);
        o.y = pack_bf16x2(a.z, a.w);
        o.z = pack_bf16x2(b.x, b.y);
        o.w = pack_bf16x2(b.z, b.w);
        out4[i] = o;
    }
}

// -------------------------------------------------------------------------
// Stage 1: hidden = silu(x @ w_in) * (x @ w_swiglu), bf16 out
// Block tile: 128(M) x 64(N) for BOTH weight matrices; 8 waves as 4x2,
// wave tile 32x32 per matrix => 8 WMMAs per K-step per wave.
// -------------------------------------------------------------------------
#define BM1 128
#define BN1 64
#define LDA 40    // padded A row (halves): 80B stride -> conflict-free frag reads
#define LDB1 72   // padded B row (halves)

__global__ __launch_bounds__(256) void gemm1_swiglu(
    const unsigned short* __restrict__ xb,     // [E][G][D] bf16
    const unsigned short* __restrict__ win,    // [E][D][F] bf16
    const unsigned short* __restrict__ wsw,    // [E][D][F] bf16
    unsigned short* __restrict__ hidden) {     // [E][G][F] bf16
    __shared__ __align__(16) unsigned short As[2][BM1 * LDA];
    __shared__ __align__(16) unsigned short Bi[2][BK * LDB1];
    __shared__ __align__(16) unsigned short Bs[2][BK * LDB1];

    const int e  = blockIdx.z;
    const int m0 = blockIdx.y * BM1;
    const int n0 = blockIdx.x * BN1;

    const unsigned short* Ag  = xb  + (size_t)e * G_ * D_ + (size_t)m0 * D_;
    const unsigned short* Wig = win + (size_t)e * D_ * F_ + n0;
    const unsigned short* Wsg = wsw + (size_t)e * D_ * F_ + n0;
    unsigned short* Hg = hidden + (size_t)e * G_ * F_;

    const int tid  = threadIdx.x;
    const int lane = tid & 31;
    const int wv   = tid >> 5;
    const int wm   = (wv >> 1) * 32;  // 0,32,64,96
    const int wn   = (wv & 1) * 32;   // 0,32

    // Cooperative copy mapping (8 bf16 = 16B chunks)
    const int ar = tid >> 2;          // A rows ar, ar+64
    const int ac = (tid & 3) * 8;
    const int br = tid >> 3;          // B row 0..31
    const int bc = (tid & 7) * 8;

    v8f accM[2][2], accG[2][2];
#pragma unroll
    for (int i = 0; i < 2; ++i)
#pragma unroll
        for (int j = 0; j < 2; ++j) { accM[i][j] = {}; accG[i][j] = {}; }

    // Prologue: async-DMA k-tile 0 into LDS buffer 0 (outside the hot loop)
    async_cp_b128(&As[0][ar * LDA + ac],        Ag + (size_t)ar * D_ + ac);
    async_cp_b128(&As[0][(ar + 64) * LDA + ac], Ag + (size_t)(ar + 64) * D_ + ac);
    async_cp_b128(&Bi[0][br * LDB1 + bc],       Wig + (size_t)br * F_ + bc);
    async_cp_b128(&Bs[0][br * LDB1 + bc],       Wsg + (size_t)br * F_ + bc);
    wait_async0();
    __syncthreads();

    const int KT = D_ / BK;
    for (int kt = 0; kt < KT; ++kt) {
        const int cur = kt & 1;
        const bool more = (kt + 1) < KT;
        v4u a0{}, a1{}, b0{}, b1{};
        if (more) {  // issue global loads early: overlap with WMMA below
            const int k = (kt + 1) * BK;
            a0 = *(const v4u*)(Ag + (size_t)ar * D_ + k + ac);
            a1 = *(const v4u*)(Ag + (size_t)(ar + 64) * D_ + k + ac);
            b0 = *(const v4u*)(Wig + (size_t)(k + br) * F_ + bc);
            b1 = *(const v4u*)(Wsg + (size_t)(k + br) * F_ + bc);
        }
        const unsigned short* at  = &As[cur][0];
        const unsigned short* bit = &Bi[cur][0];
        const unsigned short* bst = &Bs[cur][0];
        v16bf fa0 = load_a_frag(at + (wm + 0)  * LDA, LDA, lane);
        v16bf fa1 = load_a_frag(at + (wm + 16) * LDA, LDA, lane);
        v16bf fi0 = load_b_frag(bit + wn + 0,  LDB1, lane);
        v16bf fi1 = load_b_frag(bit + wn + 16, LDB1, lane);
        v16bf fs0 = load_b_frag(bst + wn + 0,  LDB1, lane);
        v16bf fs1 = load_b_frag(bst + wn + 16, LDB1, lane);
        accM[0][0] = wmma_bf16(fa0, fi0, accM[0][0]);
        accM[0][1] = wmma_bf16(fa0, fi1, accM[0][1]);
        accM[1][0] = wmma_bf16(fa1, fi0, accM[1][0]);
        accM[1][1] = wmma_bf16(fa1, fi1, accM[1][1]);
        accG[0][0] = wmma_bf16(fa0, fs0, accG[0][0]);
        accG[0][1] = wmma_bf16(fa0, fs1, accG[0][1]);
        accG[1][0] = wmma_bf16(fa1, fs0, accG[1][0]);
        accG[1][1] = wmma_bf16(fa1, fs1, accG[1][1]);
        if (more) {
            const int nxt = cur ^ 1;
            *(v4u*)(&As[nxt][ar * LDA + ac])        = a0;
            *(v4u*)(&As[nxt][(ar + 64) * LDA + ac]) = a1;
            *(v4u*)(&Bi[nxt][br * LDB1 + bc])       = b0;
            *(v4u*)(&Bs[nxt][br * LDB1 + bc])       = b1;
        }
        __syncthreads();
    }

    // Epilogue: SwiGLU, bf16 store. C/D layout: VGPR v -> M = v + 8*(lane>=16),
    // N = lane & 15.
    const int mlo = (lane >> 4) * 8;
    const int nl  = lane & 15;
#pragma unroll
    for (int ti = 0; ti < 2; ++ti) {
#pragma unroll
        for (int tj = 0; tj < 2; ++tj) {
            const int rbase = m0 + wm + ti * 16 + mlo;
            const int cbase = n0 + wn + tj * 16 + nl;
#pragma unroll
            for (int v = 0; v < 8; ++v) {
                float m = accM[ti][tj][v];
                float g = accG[ti][tj][v];
                float h = (m / (1.0f + __expf(-m))) * g;   // silu(m) * g
                Hg[(size_t)(rbase + v) * F_ + cbase] = f2bf(h);
            }
        }
    }
}

// -------------------------------------------------------------------------
// Stage 2: out = hidden @ w_out, fp32 out.
// Block tile 128x128, wave tile 32x64 (2x4 WMMA tiles), K = F = 4096.
// -------------------------------------------------------------------------
#define BM2 128
#define BN2 128
#define LDB2 136  // padded B row (halves): 272B stride

__global__ __launch_bounds__(256) void gemm2_out(
    const unsigned short* __restrict__ hidden,  // [E][G][F] bf16
    const unsigned short* __restrict__ wout,    // [E][F][D] bf16
    float* __restrict__ out) {                  // [E][G][D] f32
    __shared__ __align__(16) unsigned short As[2][BM2 * LDA];
    __shared__ __align__(16) unsigned short Bs[2][BK * LDB2];

    const int e  = blockIdx.z;
    const int m0 = blockIdx.y * BM2;
    const int n0 = blockIdx.x * BN2;

    const unsigned short* Ag = hidden + (size_t)e * G_ * F_ + (size_t)m0 * F_;
    const unsigned short* Bg = wout   + (size_t)e * F_ * D_ + n0;
    float* Og = out + (size_t)e * G_ * D_;

    const int tid  = threadIdx.x;
    const int lane = tid & 31;
    const int wv   = tid >> 5;
    const int wm   = (wv >> 1) * 32;  // 0..96
    const int wn   = (wv & 1) * 64;   // 0,64

    const int ar = tid >> 2;          // A rows ar, ar+64
    const int ac = (tid & 3) * 8;
    const int br = tid >> 4;          // B rows br, br+16
    const int bc = (tid & 15) * 8;

    v8f acc[2][4];
#pragma unroll
    for (int i = 0; i < 2; ++i)
#pragma unroll
        for (int j = 0; j < 4; ++j) acc[i][j] = {};

    // Prologue: async-DMA k-tile 0 into LDS buffer 0 (outside the hot loop)
    async_cp_b128(&As[0][ar * LDA + ac],         Ag + (size_t)ar * F_ + ac);
    async_cp_b128(&As[0][(ar + 64) * LDA + ac],  Ag + (size_t)(ar + 64) * F_ + ac);
    async_cp_b128(&Bs[0][br * LDB2 + bc],        Bg + (size_t)br * D_ + bc);
    async_cp_b128(&Bs[0][(br + 16) * LDB2 + bc], Bg + (size_t)(br + 16) * D_ + bc);
    wait_async0();
    __syncthreads();

    const int KT = F_ / BK;
    for (int kt = 0; kt < KT; ++kt) {
        const int cur = kt & 1;
        const bool more = (kt + 1) < KT;
        v4u a0{}, a1{}, b0{}, b1{};
        if (more) {
            const int k = (kt + 1) * BK;
            a0 = *(const v4u*)(Ag + (size_t)ar * F_ + k + ac);
            a1 = *(const v4u*)(Ag + (size_t)(ar + 64) * F_ + k + ac);
            b0 = *(const v4u*)(Bg + (size_t)(k + br) * D_ + bc);
            b1 = *(const v4u*)(Bg + (size_t)(k + br + 16) * D_ + bc);
        }
        const unsigned short* at = &As[cur][0];
        const unsigned short* bt = &Bs[cur][0];
        v16bf fa0 = load_a_frag(at + (wm + 0)  * LDA, LDA, lane);
        v16bf fa1 = load_a_frag(at + (wm + 16) * LDA, LDA, lane);
        v16bf fb0 = load_b_frag(bt + wn + 0,  LDB2, lane);
        v16bf fb1 = load_b_frag(bt + wn + 16, LDB2, lane);
        v16bf fb2 = load_b_frag(bt + wn + 32, LDB2, lane);
        v16bf fb3 = load_b_frag(bt + wn + 48, LDB2, lane);
        acc[0][0] = wmma_bf16(fa0, fb0, acc[0][0]);
        acc[0][1] = wmma_bf16(fa0, fb1, acc[0][1]);
        acc[0][2] = wmma_bf16(fa0, fb2, acc[0][2]);
        acc[0][3] = wmma_bf16(fa0, fb3, acc[0][3]);
        acc[1][0] = wmma_bf16(fa1, fb0, acc[1][0]);
        acc[1][1] = wmma_bf16(fa1, fb1, acc[1][1]);
        acc[1][2] = wmma_bf16(fa1, fb2, acc[1][2]);
        acc[1][3] = wmma_bf16(fa1, fb3, acc[1][3]);
        if (more) {
            const int nxt = cur ^ 1;
            *(v4u*)(&As[nxt][ar * LDA + ac])         = a0;
            *(v4u*)(&As[nxt][(ar + 64) * LDA + ac])  = a1;
            *(v4u*)(&Bs[nxt][br * LDB2 + bc])        = b0;
            *(v4u*)(&Bs[nxt][(br + 16) * LDB2 + bc]) = b1;
        }
        __syncthreads();
    }

    const int mlo = (lane >> 4) * 8;
    const int nl  = lane & 15;
#pragma unroll
    for (int ti = 0; ti < 2; ++ti) {
#pragma unroll
        for (int tj = 0; tj < 4; ++tj) {
            const int rbase = m0 + wm + ti * 16 + mlo;
            const int cbase = n0 + wn + tj * 16 + nl;
#pragma unroll
            for (int v = 0; v < 8; ++v) {
                Og[(size_t)(rbase + v) * D_ + cbase] = acc[ti][tj][v];
            }
        }
    }
}

// -------------------------------------------------------------------------
extern "C" void kernel_launch(void* const* d_in, const int* in_sizes, int n_in,
                              void* d_out, int out_size, void* d_ws, size_t ws_size,
                              hipStream_t stream) {
    const float* x    = (const float*)d_in[0];  // [E*G, D]
    const float* win  = (const float*)d_in[1];  // [E*D, F]
    const float* wsw  = (const float*)d_in[2];  // [E*D, F]
    const float* wout = (const float*)d_in[3];  // [E*F, D]
    float* out = (float*)d_out;

    // Workspace layout (bf16 halves): x | w_in | w_swiglu | w_out | hidden
    unsigned short* xb    = (unsigned short*)d_ws;
    unsigned short* winb  = xb    + (size_t)E_ * G_ * D_;
    unsigned short* wswb  = winb  + (size_t)E_ * D_ * F_;
    unsigned short* woutb = wswb  + (size_t)E_ * D_ * F_;
    unsigned short* hid   = woutb + (size_t)E_ * F_ * D_;
    const size_t need = ((size_t)E_ * G_ * D_ + 3ull * E_ * D_ * F_ +
                         (size_t)E_ * G_ * F_) * sizeof(unsigned short);
    if (ws_size < need) return;  // workspace too small: nothing sane to do

    cvt_bf16<<<2048, 256, 0, stream>>>(x,    xb,    (long long)E_ * G_ * D_ / 8);
    cvt_bf16<<<4096, 256, 0, stream>>>(win,  winb,  (long long)E_ * D_ * F_ / 8);
    cvt_bf16<<<4096, 256, 0, stream>>>(wsw,  wswb,  (long long)E_ * D_ * F_ / 8);
    cvt_bf16<<<4096, 256, 0, stream>>>(wout, woutb, (long long)E_ * F_ * D_ / 8);

    dim3 g1(F_ / BN1, G_ / BM1, E_);  // (64, 16, 8)
    gemm1_swiglu<<<g1, 256, 0, stream>>>(xb, winb, wswb, hid);

    dim3 g2(D_ / BN2, G_ / BM2, E_);  // (8, 16, 8)
    gemm2_out<<<g2, 256, 0, stream>>>(hid, woutb, out);
}
